// CausalSelfAttention_89575837925751
// MI455X (gfx1250) — matrix operand order repaired and checked by reading, not compiled, not run on hardware
//
#include <hip/hip_runtime.h>

// ---------------------------------------------------------------------------
// CausalSelfAttention forward for MI455X (gfx1250, wave32, WMMA bf16)
// B=2, S=2048, D=1024, H=16, dh=64
// Pipeline: f32->bf16 convert | QKV GEMMs (wmma bf16, async-LDS staging) |
//           flash attention (wmma bf16, online softmax, TDM tile staging) |
//           output projection GEMM (f32 out)
// ---------------------------------------------------------------------------

typedef __attribute__((ext_vector_type(16))) __bf16          v16bf;
typedef __attribute__((ext_vector_type(8)))  float           v8f;
typedef __attribute__((ext_vector_type(16))) unsigned short  v16us;
typedef __attribute__((ext_vector_type(4)))  unsigned int    uint4v;
typedef __attribute__((ext_vector_type(4)))  int             int4v;
typedef __attribute__((ext_vector_type(8)))  int             int8v;

#define DMODEL 1024
#define NHEADS 16
#define DHEAD  64
#define BATCH  2
#define SEQ    2048
#define MTOT   (BATCH * SEQ)

// ---- gfx1250 feature detection (guarded; plain fallbacks) -----------------
#if defined(__has_builtin)
#if __has_builtin(__builtin_amdgcn_global_load_async_to_lds_b128)
#define HAVE_ASYNC_LDS 1
#endif
#if __has_builtin(__builtin_amdgcn_s_wait_asynccnt)
#define HAVE_WAIT_ASYNC 1
#endif
#if __has_builtin(__builtin_amdgcn_tensor_load_to_lds)
#define HAVE_TDM 1
#endif
#if __has_builtin(__builtin_amdgcn_s_wait_tensorcnt)
#define HAVE_WAIT_TENSOR 1
#endif
#endif

// addrspace(1)=global ("__device__"), addrspace(3)=LDS
typedef __attribute__((address_space(1))) int4v* g4_ptr;
typedef __attribute__((address_space(3))) int4v* l4_ptr;

// Copy 16 bytes global -> LDS (async on gfx1250 when available)
static __device__ __forceinline__ void copy16_g2l(const void* g, void* l) {
#ifdef HAVE_ASYNC_LDS
    __builtin_amdgcn_global_load_async_to_lds_b128((g4_ptr)g, (l4_ptr)l, 0, 0);
#else
    *(uint4v*)l = *(const uint4v*)g;
#endif
}

static __device__ __forceinline__ void async_wait_all() {
#ifdef HAVE_ASYNC_LDS
#ifdef HAVE_WAIT_ASYNC
    __builtin_amdgcn_s_wait_asynccnt(0);
#else
    asm volatile("s_wait_asynccnt 0" ::: "memory");
#endif
#endif
}

#ifdef HAVE_TDM
static __device__ __forceinline__ void tensor_wait_all() {
#ifdef HAVE_WAIT_TENSOR
    __builtin_amdgcn_s_wait_tensorcnt(0);
#else
    asm volatile("s_wait_tensorcnt 0" ::: "memory");
#endif
}

// TDM: load a 64x64 bf16 tile (rows stride DMODEL elements) from a
// [MTOT, DMODEL] bf16 tensor into LDS rows of 72 shorts (128B data + 16B pad).
// D# per CDNA5 ISA ch.8: group0 {count, lds_addr, global_addr, type=2},
// group1 {data_size=2B, pad_enable, pad_interval=32dw(128B), pad_amount=4dw,
//         tensor_dim0/1, tile_dim0/1, tensor_dim0_stride}, groups 2/3 = 0 (2D).
static __device__ __forceinline__ void tdm_load_tile64x64(void* lds, const void* g) {
    unsigned ldsOff = (unsigned)(unsigned long long)(l4_ptr)lds;   // LDS byte offset
    unsigned long long ga = (unsigned long long)(size_t)g;         // global byte addr
    uint4v g0;
    g0[0] = 1u;                                                    // count=1 (valid)
    g0[1] = ldsOff;                                                // lds_addr
    g0[2] = (unsigned)(ga & 0xFFFFFFFFull);                        // global_addr[31:0]
    g0[3] = ((unsigned)(ga >> 32) & 0x01FFFFFFu) | (2u << 30);     // [56:32] | type=2
    int8v g1;
    g1[0] = (int)((1u << 16)        // data_size = 1 -> 2 bytes
                | (1u << 20)        // pad_enable
                | (4u << 22)        // pad_interval: 2^4*8B = 128B (32 dwords)
                | (3u << 25));      // pad_amount: 4 dwords = 16B
    g1[1] = (int)(((unsigned)DMODEL & 0xFFFFu) << 16);             // tensor_dim0 lo
    g1[2] = (int)((((unsigned)MTOT & 0xFFFFu) << 16) | ((unsigned)DMODEL >> 16));
    g1[3] = (int)((64u << 16) | ((unsigned)MTOT >> 16));           // tile_dim0 | dim1 hi
    g1[4] = 64;                                                    // tile_dim1 (dim2=0)
    g1[5] = DMODEL;                                                // dim0_stride lo
    g1[6] = 0;
    g1[7] = 0;
    int4v gz = (int4v){0, 0, 0, 0};
#if defined(__clang_major__) && __clang_major__ >= 23
    int8v gz8 = (int8v){0, 0, 0, 0, 0, 0, 0, 0};
    __builtin_amdgcn_tensor_load_to_lds(g0, g1, gz, gz, gz8, 0);
#else
    __builtin_amdgcn_tensor_load_to_lds(g0, g1, gz, gz, 0);
#endif
}
#endif  // HAVE_TDM

static __device__ __forceinline__ unsigned short f32_to_bf16(float f) {
    unsigned u = __builtin_bit_cast(unsigned, f);
    unsigned r = u + 0x7FFFu + ((u >> 16) & 1u);   // round-to-nearest-even
    return (unsigned short)(r >> 16);
}

// K-index inside a 32-wide chunk for 16-bit A-matrix fragments (contiguous, matches B):
// VGPR v (pair 2v,2v+1): K = 16*half + 2v
static __device__ __forceinline__ int a_kk(int v, int half) {
    return 16 * half + 2 * v;
}

// ---------------------------------------------------------------------------
// Kernel 0: f32 -> bf16 conversion
// ---------------------------------------------------------------------------
__global__ void f32_to_bf16_kernel(const float* __restrict__ src,
                                   unsigned short* __restrict__ dst, int n) {
    int i = blockIdx.x * blockDim.x + threadIdx.x;
    if (i < n) dst[i] = f32_to_bf16(src[i]);
}

// ---------------------------------------------------------------------------
// Kernel 1: tiled bf16 GEMM  out = (A[MxK] * W[KxN] + bias) * scale
// 64x64 block tile, 128 threads = 4 waves, wave -> 32x32 (2x2 WMMA accs)
// A tile async-staged row-major; W tile staged transposed ([n][k]) so both
// A and B fragment reads are contiguous ds_load_b128 pairs.
// ---------------------------------------------------------------------------
template <bool OUT_BF16>
__global__ __launch_bounds__(128)
void gemm_bf16_kernel(const unsigned short* __restrict__ A,
                      const unsigned short* __restrict__ W,
                      const float* __restrict__ bias,
                      void* __restrict__ out,
                      int M, int N, int Kd, float scale) {
    __shared__ unsigned short Asl[64][40];    // 64 rows x 32 k (+pad)
    __shared__ unsigned short Bslt[64][40];   // 64 n x 32 k (transposed, +pad)

    const int t    = threadIdx.x;
    const int lane = t & 31;
    const int wave = t >> 5;
    const int wm   = wave >> 1;      // 0..1
    const int wn   = wave & 1;       // 0..1
    const int half = lane >> 4;
    const int lm   = lane & 15;

    const int row0 = blockIdx.y * 64;
    const int col0 = blockIdx.x * 64;

    v8f c[2][2];
#pragma unroll
    for (int i = 0; i < 2; ++i)
#pragma unroll
        for (int j = 0; j < 2; ++j)
            c[i][j] = (v8f){0.f, 0.f, 0.f, 0.f, 0.f, 0.f, 0.f, 0.f};

    for (int k0 = 0; k0 < Kd; k0 += 32) {
        // Stage A tile (async): 64 rows x 32 bf16; 2 threads/row, 2x16B each
        {
            int r = t >> 1, s = t & 1;
            const unsigned short* g = A + (size_t)(row0 + r) * Kd + k0 + s * 16;
            copy16_g2l(g,     &Asl[r][s * 16]);
            copy16_g2l(g + 8, &Asl[r][s * 16 + 8]);
            if (k0 + 32 < Kd)
                __builtin_prefetch(g + 32, 0, 0);
        }
        // Stage W tile transposed: read 16 contiguous cols of one k-row,
        // scatter into Bslt[n][k]
        {
            int r = t >> 2, s = t & 3;
            const uint4v* src = (const uint4v*)(W + (size_t)(k0 + r) * N + col0 + s * 16);
            union { uint4v v[2]; unsigned short us[16]; } wt;
            wt.v[0] = src[0];
            wt.v[1] = src[1];
#pragma unroll
            for (int j = 0; j < 16; ++j) Bslt[s * 16 + j][r] = wt.us[j];
        }
        async_wait_all();
        __syncthreads();

        v16us au[2], bu[2];
#pragma unroll
        for (int mb = 0; mb < 2; ++mb) {
            int row = wm * 32 + mb * 16 + lm;
#pragma unroll
            for (int v = 0; v < 8; ++v) {
                int kk = a_kk(v, half);
                unsigned pair = *(const unsigned*)(&Asl[row][kk]);
                au[mb][2 * v]     = (unsigned short)(pair & 0xFFFFu);
                au[mb][2 * v + 1] = (unsigned short)(pair >> 16);
            }
        }
#pragma unroll
        for (int nb = 0; nb < 2; ++nb) {
            int col = wn * 32 + nb * 16 + lm;
#pragma unroll
            for (int v = 0; v < 8; ++v) {
                // B 32x16: half-wave selects K 0-15 / 16-31; contiguous in Bslt row
                unsigned pair = *(const unsigned*)(&Bslt[col][half * 16 + 2 * v]);
                bu[nb][2 * v]     = (unsigned short)(pair & 0xFFFFu);
                bu[nb][2 * v + 1] = (unsigned short)(pair >> 16);
            }
        }
#pragma unroll
        for (int mb = 0; mb < 2; ++mb)
#pragma unroll
            for (int nb = 0; nb < 2; ++nb)
                c[mb][nb] = __builtin_amdgcn_wmma_f32_16x16x32_bf16(
                    false, __builtin_bit_cast(v16bf, au[mb]),
                    false, __builtin_bit_cast(v16bf, bu[nb]),
                    (short)0, c[mb][nb], false, false);
        __syncthreads();
    }

    // Epilogue: C layout VGPR v -> row M = v + 8*half, col N = lane%16
#pragma unroll
    for (int mb = 0; mb < 2; ++mb)
#pragma unroll
        for (int nb = 0; nb < 2; ++nb)
#pragma unroll
            for (int v = 0; v < 8; ++v) {
                int row = row0 + wm * 32 + mb * 16 + v + 8 * half;
                int col = col0 + wn * 32 + nb * 16 + lm;
                float val = (c[mb][nb][v] + bias[col]) * scale;
                if (OUT_BF16)
                    ((unsigned short*)out)[(size_t)row * N + col] = f32_to_bf16(val);
                else
                    ((float*)out)[(size_t)row * N + col] = val;
            }
}

// ---------------------------------------------------------------------------
// Kernel 2: flash attention (causal + padding mask), Q pre-scaled by 1/sqrt(dh)
// One block per (64-query tile, head, batch); 4 waves, wave -> 16 q rows.
// K/V tiles staged by the Tensor Data Mover (TDM) when available.
// ---------------------------------------------------------------------------
__global__ __launch_bounds__(128)
void flash_attn_kernel(const unsigned short* __restrict__ Qg,
                       const unsigned short* __restrict__ Kg,
                       const unsigned short* __restrict__ Vg,
                       const float* __restrict__ pmask,
                       unsigned short* __restrict__ Og) {
    __shared__ unsigned short Ksl[64][72];       // 64 keys x 64 depth (+16B pad)
    __shared__ unsigned short Vsl[64][72];
    __shared__ unsigned short Psl[4][16][72];    // per-wave P tile 16x64
    __shared__ float pml[64];

    const int t    = threadIdx.x;
    const int lane = t & 31;
    const int w    = t >> 5;
    const int half = lane >> 4;
    const int lm   = lane & 15;

    const int q0 = blockIdx.x * 64;
    const int h  = blockIdx.y;
    const int b  = blockIdx.z;

    const size_t baseRow = (size_t)b * SEQ;
    const int hc = h * DHEAD;

    // Q A-fragments: 16 rows x 64 depth = 2 chunks of K=32
    v16us qa[2];
    const int qrow = q0 + w * 16 + lm;
#pragma unroll
    for (int kc = 0; kc < 2; ++kc)
#pragma unroll
        for (int v = 0; v < 8; ++v) {
            int d = kc * 32 + a_kk(v, half);
            unsigned pair =
                *(const unsigned*)(Qg + (baseRow + qrow) * DMODEL + hc + d);
            qa[kc][2 * v]     = (unsigned short)(pair & 0xFFFFu);
            qa[kc][2 * v + 1] = (unsigned short)(pair >> 16);
        }

    v8f o[4];
#pragma unroll
    for (int i = 0; i < 4; ++i)
        o[i] = (v8f){0.f, 0.f, 0.f, 0.f, 0.f, 0.f, 0.f, 0.f};
    float mrun[8], lrun[8];
#pragma unroll
    for (int v = 0; v < 8; ++v) { mrun[v] = -3.0e38f; lrun[v] = 0.f; }

    const int kbmax = q0 / 64;          // causal: only past + diagonal blocks
    for (int kb = 0; kb <= kbmax; ++kb) {
#ifdef HAVE_TDM
        // TDM: wave 0 issues two 64x64 tile DMAs (K and V) and waits TENSORcnt;
        // barrier publishes the LDS tiles to all waves.
        if (t < 64) pml[t] = pmask[b * SEQ + kb * 64 + t];
        if (w == 0) {
            tdm_load_tile64x64(&Ksl[0][0], Kg + (baseRow + kb * 64) * DMODEL + hc);
            tdm_load_tile64x64(&Vsl[0][0], Vg + (baseRow + kb * 64) * DMODEL + hc);
            tensor_wait_all();
        }
#else
        // Fallback: async global->LDS; threads 0..63 K rows, 64..127 V rows.
        if (t < 64) {
            const unsigned short* g = Kg + (baseRow + kb * 64 + t) * DMODEL + hc;
#pragma unroll
            for (int i = 0; i < 8; ++i)
                copy16_g2l(g + i * 8, &Ksl[t][i * 8]);
            pml[t] = pmask[b * SEQ + kb * 64 + t];
        } else {
            int r = t - 64;
            const unsigned short* g = Vg + (baseRow + kb * 64 + r) * DMODEL + hc;
#pragma unroll
            for (int i = 0; i < 8; ++i)
                copy16_g2l(g + i * 8, &Vsl[r][i * 8]);
        }
        async_wait_all();
#endif
        __syncthreads();

        // S = Q * K^T  (16 x 64 per wave); B[d][key] = K[key][d] = Ksl[key][d]
        v8f s[4];
#pragma unroll
        for (int i = 0; i < 4; ++i)
            s[i] = (v8f){0.f, 0.f, 0.f, 0.f, 0.f, 0.f, 0.f, 0.f};
#pragma unroll
        for (int kc = 0; kc < 2; ++kc) {
            v16bf a = __builtin_bit_cast(v16bf, qa[kc]);
#pragma unroll
            for (int nch = 0; nch < 4; ++nch) {
                v16us bu;
#pragma unroll
                for (int v = 0; v < 8; ++v) {
                    int kk = kc * 32 + half * 16 + 2 * v;
                    unsigned pair = *(const unsigned*)(&Ksl[nch * 16 + lm][kk]);
                    bu[2 * v]     = (unsigned short)(pair & 0xFFFFu);
                    bu[2 * v + 1] = (unsigned short)(pair >> 16);
                }
                s[nch] = __builtin_amdgcn_wmma_f32_16x16x32_bf16(
                    false, a, false, __builtin_bit_cast(v16bf, bu),
                    (short)0, s[nch], false, false);
            }
        }

        // Online softmax. Row M = v + 8*half; reductions inside 16-lane halves.
#pragma unroll
        for (int v = 0; v < 8; ++v) {
            int qg = q0 + w * 16 + v + 8 * half;
            float sv[4];
            float mx = -3.0e38f;
#pragma unroll
            for (int tn = 0; tn < 4; ++tn) {
                int key = kb * 64 + tn * 16 + lm;
                float x = s[tn][v];
                x += (1.0f - pml[tn * 16 + lm]) * -1.0e9f;   // padding mask
                if (key > qg) x = -1.0e9f;                    // causal mask
                sv[tn] = x;
                mx = fmaxf(mx, x);
            }
            mx = fmaxf(mx, __shfl_xor(mx, 1, 32));
            mx = fmaxf(mx, __shfl_xor(mx, 2, 32));
            mx = fmaxf(mx, __shfl_xor(mx, 4, 32));
            mx = fmaxf(mx, __shfl_xor(mx, 8, 32));
            float mnew = fmaxf(mrun[v], mx);
            float corr = __expf(mrun[v] - mnew);
            float ls = 0.f;
#pragma unroll
            for (int tn = 0; tn < 4; ++tn) {
                float p = __expf(sv[tn] - mnew);
                ls += p;
                Psl[w][v + 8 * half][tn * 16 + lm] = f32_to_bf16(p);
            }
            ls += __shfl_xor(ls, 1, 32);
            ls += __shfl_xor(ls, 2, 32);
            ls += __shfl_xor(ls, 4, 32);
            ls += __shfl_xor(ls, 8, 32);
            lrun[v] = lrun[v] * corr + ls;
            mrun[v] = mnew;
#pragma unroll
            for (int nch = 0; nch < 4; ++nch) o[nch][v] *= corr;
        }
        // Per-wave P tile: same-wave DS ops are in order; no barrier needed.

        // O += P * V   (A = P 16x64, B = V keys x depth)
#pragma unroll
        for (int kc = 0; kc < 2; ++kc) {
            v16us pu;
#pragma unroll
            for (int v = 0; v < 8; ++v) {
                int kk = kc * 32 + a_kk(v, half);
                unsigned pair = *(const unsigned*)(&Psl[w][lm][kk]);
                pu[2 * v]     = (unsigned short)(pair & 0xFFFFu);
                pu[2 * v + 1] = (unsigned short)(pair >> 16);
            }
            v16bf a = __builtin_bit_cast(v16bf, pu);
#pragma unroll
            for (int nch = 0; nch < 4; ++nch) {
                v16us bu;
#pragma unroll
                for (int v = 0; v < 8; ++v) {
                    int kk = kc * 32 + half * 16 + 2 * v;
                    bu[2 * v]     = Vsl[kk][nch * 16 + lm];
                    bu[2 * v + 1] = Vsl[kk + 1][nch * 16 + lm];
                }
                o[nch] = __builtin_amdgcn_wmma_f32_16x16x32_bf16(
                    false, a, false, __builtin_bit_cast(v16bf, bu),
                    (short)0, o[nch], false, false);
            }
        }
        __syncthreads();
    }

    // Finalize: O /= l ; store bf16 into (B,S,H*dh) layout
#pragma unroll
    for (int v = 0; v < 8; ++v) {
        float inv = lrun[v] > 0.f ? 1.0f / lrun[v] : 0.f;
        int row = q0 + w * 16 + v + 8 * half;
#pragma unroll
        for (int nch = 0; nch < 4; ++nch) {
            int col = hc + nch * 16 + lm;
            Og[(baseRow + row) * DMODEL + col] = f32_to_bf16(o[nch][v] * inv);
        }
    }
}

// ---------------------------------------------------------------------------
// Host-side launch
// ---------------------------------------------------------------------------
extern "C" void kernel_launch(void* const* d_in, const int* in_sizes, int n_in,
                              void* d_out, int out_size, void* d_ws, size_t ws_size,
                              hipStream_t stream) {
    const float* x   = (const float*)d_in[0];
    const float* pm  = (const float*)d_in[1];
    const float* wqw = (const float*)d_in[2];
    const float* wqb = (const float*)d_in[3];
    const float* wkw = (const float*)d_in[4];
    const float* wkb = (const float*)d_in[5];
    const float* wvw = (const float*)d_in[6];
    const float* wvb = (const float*)d_in[7];
    const float* wow = (const float*)d_in[8];
    const float* wob = (const float*)d_in[9];
    float* out = (float*)d_out;

    const size_t SZ_ACT = (size_t)MTOT * DMODEL * sizeof(unsigned short);   // 8 MiB
    const size_t SZ_W   = (size_t)DMODEL * DMODEL * sizeof(unsigned short); // 2 MiB

    char* ws = (char*)d_ws;
    unsigned short* xb  = (unsigned short*)(ws);                 // reused for O
    unsigned short* wqf = (unsigned short*)(ws + SZ_ACT);
    unsigned short* wkf = (unsigned short*)(ws + SZ_ACT + SZ_W);
    unsigned short* wvf = (unsigned short*)(ws + SZ_ACT + 2 * SZ_W);
    unsigned short* wof = (unsigned short*)(ws + SZ_ACT + 3 * SZ_W);
    unsigned short* Qb  = (unsigned short*)(ws + SZ_ACT + 4 * SZ_W);
    unsigned short* Kb  = (unsigned short*)(ws + 2 * SZ_ACT + 4 * SZ_W);
    unsigned short* Vb  = (unsigned short*)(ws + 3 * SZ_ACT + 4 * SZ_W);
    unsigned short* Ob  = xb;   // xb dead after QKV GEMMs -> reuse for attn out

    const int nAct = MTOT * DMODEL;
    const int nW   = DMODEL * DMODEL;
    f32_to_bf16_kernel<<<(nAct + 255) / 256, 256, 0, stream>>>(x,   xb,  nAct);
    f32_to_bf16_kernel<<<(nW   + 255) / 256, 256, 0, stream>>>(wqw, wqf, nW);
    f32_to_bf16_kernel<<<(nW   + 255) / 256, 256, 0, stream>>>(wkw, wkf, nW);
    f32_to_bf16_kernel<<<(nW   + 255) / 256, 256, 0, stream>>>(wvw, wvf, nW);
    f32_to_bf16_kernel<<<(nW   + 255) / 256, 256, 0, stream>>>(wow, wof, nW);

    dim3 gg(DMODEL / 64, MTOT / 64);
    // Q scaled by 1/sqrt(dh) = 0.125 so attention logits need no extra scale
    gemm_bf16_kernel<true><<<gg, 128, 0, stream>>>(xb, wqf, wqb, Qb, MTOT, DMODEL, DMODEL, 0.125f);
    gemm_bf16_kernel<true><<<gg, 128, 0, stream>>>(xb, wkf, wkb, Kb, MTOT, DMODEL, DMODEL, 1.0f);
    gemm_bf16_kernel<true><<<gg, 128, 0, stream>>>(xb, wvf, wvb, Vb, MTOT, DMODEL, DMODEL, 1.0f);

    dim3 fg(SEQ / 64, NHEADS, BATCH);
    flash_attn_kernel<<<fg, 128, 0, stream>>>(Qb, Kb, Vb, pm, Ob);

    gemm_bf16_kernel<false><<<gg, 128, 0, stream>>>(Ob, wof, wob, (void*)out, MTOT, DMODEL, DMODEL, 1.0f);
}
